// PointNet2_ORI_2619930051204
// MI455X (gfx1250) — compile-verified, matrix operand order checked by reading, not run on hardware
//
#include <hip/hip_runtime.h>
#include <hip/hip_bf16.h>

// ---------------------------------------------------------------------------
// Types for CDNA5 WMMA (gfx1250, wave32)
// ---------------------------------------------------------------------------
typedef __bf16 v16bf __attribute__((ext_vector_type(16)));
typedef float  v8f   __attribute__((ext_vector_type(8)));

union FragAB {
    v16bf v;
    uint4 q[2];
};

__device__ __forceinline__ unsigned short f2bf(float f) {
    unsigned int u = __builtin_bit_cast(unsigned int, f);
    unsigned int r = 0x7FFFu + ((u >> 16) & 1u);   // round-to-nearest-even
    u += r;
    return (unsigned short)(u >> 16);
}

#define CDIV(a, b) (((a) + (b) - 1) / (b))

// ---------------------------------------------------------------------------
// Weight prep: f32 [Cout,Cin] -> bf16 [CoutPad,KPad] (zero padded),
// BN folded into per-channel scale/bias.
// ---------------------------------------------------------------------------
__global__ void prep_w_kernel(const float* __restrict__ w, int Cout, int Cin,
                              int CoutPad, int KPad, unsigned short* __restrict__ Wb) {
    int i = blockIdx.x * blockDim.x + threadIdx.x;
    int total = CoutPad * KPad;
    if (i >= total) return;
    int co = i / KPad, ci = i % KPad;
    float v = (co < Cout && ci < Cin) ? w[(size_t)co * Cin + ci] : 0.0f;
    Wb[i] = f2bf(v);
}

__global__ void prep_sb_kernel(const float* __restrict__ b, const float* __restrict__ gamma,
                               const float* __restrict__ beta, const float* __restrict__ mean,
                               const float* __restrict__ var, int Cout, int CoutPad,
                               int has_bn, float* __restrict__ scale, float* __restrict__ bias) {
    int c = blockIdx.x * blockDim.x + threadIdx.x;
    if (c >= CoutPad) return;
    if (c >= Cout) { scale[c] = 0.0f; bias[c] = 0.0f; return; }
    if (has_bn) {
        float s = gamma[c] * rsqrtf(var[c] + 1e-5f);
        scale[c] = s;
        bias[c]  = (b[c] - mean[c]) * s + beta[c];
    } else {
        scale[c] = 1.0f;
        bias[c]  = b[c];
    }
}

// ---------------------------------------------------------------------------
// Transpose xyz [B,3,N] -> [B,N,3]
// ---------------------------------------------------------------------------
__global__ void transpose_xyz_kernel(const float* __restrict__ in, int B, int N,
                                     float* __restrict__ out) {
    int i = blockIdx.x * blockDim.x + threadIdx.x;
    if (i >= B * N) return;
    int b = i / N, n = i % N;
    out[(size_t)i * 3 + 0] = in[((size_t)b * 3 + 0) * N + n];
    out[(size_t)i * 3 + 1] = in[((size_t)b * 3 + 1) * N + n];
    out[(size_t)i * 3 + 2] = in[((size_t)b * 3 + 2) * N + n];
}

// ---------------------------------------------------------------------------
// Farthest point sampling: one block per batch, min-dist array in LDS.
// ---------------------------------------------------------------------------
__global__ void fps_kernel(const float* __restrict__ xyz, int N, int npoint,
                           int* __restrict__ out) {
    int b = blockIdx.x;
    const float* p = xyz + (size_t)b * N * 3;
    __shared__ float dist[4096];
    __shared__ float rmax[256];
    __shared__ int   ridx[256];
    __shared__ int   far_s;
    int tid = threadIdx.x;
    for (int n = tid; n < N; n += blockDim.x) dist[n] = 1e10f;
    if (tid == 0) far_s = 0;
    __syncthreads();
    for (int i = 0; i < npoint; i++) {
        int far = far_s;
        if (tid == 0) out[b * npoint + i] = far;
        float cx = p[far * 3 + 0], cy = p[far * 3 + 1], cz = p[far * 3 + 2];
        float bm = -1.0f; int bi = 0;
        for (int n = tid; n < N; n += blockDim.x) {
            float dx = p[n * 3 + 0] - cx;
            float dy = p[n * 3 + 1] - cy;
            float dz = p[n * 3 + 2] - cz;
            float d = dx * dx + dy * dy + dz * dz;
            float dm = fminf(dist[n], d);
            dist[n] = dm;
            if (dm > bm) { bm = dm; bi = n; }
        }
        rmax[tid] = bm; ridx[tid] = bi;
        __syncthreads();
        for (int s = blockDim.x / 2; s > 0; s >>= 1) {
            if (tid < s) {
                if (rmax[tid + s] > rmax[tid] ||
                    (rmax[tid + s] == rmax[tid] && ridx[tid + s] < ridx[tid])) {
                    rmax[tid] = rmax[tid + s];
                    ridx[tid] = ridx[tid + s];
                }
            }
            __syncthreads();
        }
        if (tid == 0) far_s = ridx[0];
        __syncthreads();
    }
}

// ---------------------------------------------------------------------------
// Gather sampled centers: new_xyz[b,s,:] = xyz[b, idx[b,s], :]
// ---------------------------------------------------------------------------
__global__ void gather_xyz_kernel(const float* __restrict__ xyz, const int* __restrict__ idx,
                                  int B, int N, int S, float* __restrict__ out) {
    int t = blockIdx.x * blockDim.x + threadIdx.x;
    if (t >= B * S) return;
    int b = t / S;
    int id = idx[t];
    const float* g = xyz + ((size_t)b * N + id) * 3;
    out[(size_t)t * 3 + 0] = g[0];
    out[(size_t)t * 3 + 1] = g[1];
    out[(size_t)t * 3 + 2] = g[2];
}

// ---------------------------------------------------------------------------
// Ball query: ordered scan == reference's sort-then-take-first-nsample.
// ---------------------------------------------------------------------------
__global__ void ball_query_kernel(const float* __restrict__ xyz, const float* __restrict__ nxyz,
                                  int B, int N, int S, float r2, int nsample,
                                  int* __restrict__ idx) {
    int t = blockIdx.x * blockDim.x + threadIdx.x;
    if (t >= B * S) return;
    int b = t / S;
    const float* p = xyz + (size_t)b * N * 3;
    float cx = nxyz[(size_t)t * 3 + 0];
    float cy = nxyz[(size_t)t * 3 + 1];
    float cz = nxyz[(size_t)t * 3 + 2];
    int* o = idx + (size_t)t * nsample;
    int cnt = 0;
    for (int n = 0; n < N && cnt < nsample; n++) {
        float dx = p[n * 3 + 0] - cx;
        float dy = p[n * 3 + 1] - cy;
        float dz = p[n * 3 + 2] - cz;
        float d = dx * dx + dy * dy + dz * dz;
        if (d <= r2) o[cnt++] = n;
    }
    int f = (cnt > 0) ? o[0] : 0;
    for (int j = cnt; j < nsample; j++) o[j] = f;
}

// ---------------------------------------------------------------------------
// Grouping: build padded bf16 rows [gnorm(3) | gxyz(3) | pts(C) | 0...]
// ---------------------------------------------------------------------------
__global__ void group_kernel(const float* __restrict__ xyz, const float* __restrict__ nxyz,
                             const int* __restrict__ idx, const float* __restrict__ pts,
                             int B, int Nsrc, int S, int K, int C, int KPad,
                             unsigned short* __restrict__ X) {
    int t = blockIdx.x * blockDim.x + threadIdx.x;
    if (t >= B * S * K) return;
    int s = (t / K) % S;
    int b = t / (K * S);
    int id = idx[t];
    const float* g  = xyz  + ((size_t)b * Nsrc + id) * 3;
    const float* c0 = nxyz + ((size_t)b * S + s) * 3;
    unsigned short* xo = X + (size_t)t * KPad;
    float gx = g[0], gy = g[1], gz = g[2];
    xo[0] = f2bf(gx - c0[0]);
    xo[1] = f2bf(gy - c0[1]);
    xo[2] = f2bf(gz - c0[2]);
    xo[3] = f2bf(gx);
    xo[4] = f2bf(gy);
    xo[5] = f2bf(gz);
    int w = 6;
    if (pts) {
        const float* pp = pts + ((size_t)b * Nsrc + id) * C;
        for (int c = 0; c < C; c++) xo[w++] = f2bf(pp[c]);
    }
    for (; w < KPad; w++) xo[w] = 0;
}

// group_all: rows [xyz(3) | pts(C) | 0...]
__global__ void group_all_kernel(const float* __restrict__ xyz, const float* __restrict__ pts,
                                 int B, int Nsrc, int C, int KPad,
                                 unsigned short* __restrict__ X) {
    int t = blockIdx.x * blockDim.x + threadIdx.x;
    if (t >= B * Nsrc) return;
    const float* g = xyz + (size_t)t * 3;
    unsigned short* xo = X + (size_t)t * KPad;
    xo[0] = f2bf(g[0]);
    xo[1] = f2bf(g[1]);
    xo[2] = f2bf(g[2]);
    int w = 3;
    const float* pp = pts + (size_t)t * C;
    for (int c = 0; c < C; c++) xo[w++] = f2bf(pp[c]);
    for (; w < KPad; w++) xo[w] = 0;
}

// ---------------------------------------------------------------------------
// Fused GEMM + BN + activation, bf16 WMMA, f32 accumulate.
// Block = 128 threads = 4 waves; wave computes a 32x32 tile via 2x2 WMMA frags.
// X: [M, KPad] bf16 row-major.  W: [CoutPad, KPad] bf16 row-major.
// act: 0 = none, 1 = relu, 2 = leaky(0.2).
//
// NOTE: A-fragment loads are UNGUARDED. Caller guarantees the X buffer is
// readable for ceil(M/128)*128 rows (activation slots are 64 MB). Phantom
// rows only feed phantom output rows, which the guarded epilogue discards.
// This keeps the inner loop free of EXEC manipulation (WMMA wants EXEC=all 1s).
// ---------------------------------------------------------------------------
__global__ void __launch_bounds__(128) gemm_bn_act_kernel(
    const unsigned short* __restrict__ X, int M, int KPad,
    const unsigned short* __restrict__ W,
    const float* __restrict__ scale, const float* __restrict__ bias,
    float* __restrict__ Yf, unsigned short* __restrict__ Yb, int ldy, int act) {

    int lane = threadIdx.x & 31;
    int wid  = threadIdx.x >> 5;
    int mbase = blockIdx.x * 128 + wid * 32;
    int nbase = blockIdx.y * 32;
    int lr = lane & 15;
    int kh = (lane >> 4) << 3;   // 0 or 8: K-half per ISA 16-bit A/B layout

    v8f acc[2][2] = {};

    const unsigned short* pA0 = X + (size_t)(mbase + lr) * KPad + kh;
    const unsigned short* pA1 = X + (size_t)(mbase + 16 + lr) * KPad + kh;
    const unsigned short* pB0 = W + (size_t)(nbase + lr) * KPad + kh;
    const unsigned short* pB1 = W + (size_t)(nbase + 16 + lr) * KPad + kh;

    for (int k = 0; k < KPad; k += 32) {
        FragAB a0, a1, b0, b1;
        a0.q[0] = *(const uint4*)(pA0 + k);
        a0.q[1] = *(const uint4*)(pA0 + k + 16);
        a1.q[0] = *(const uint4*)(pA1 + k);
        a1.q[1] = *(const uint4*)(pA1 + k + 16);
        b0.q[0] = *(const uint4*)(pB0 + k);
        b0.q[1] = *(const uint4*)(pB0 + k + 16);
        b1.q[0] = *(const uint4*)(pB1 + k);
        b1.q[1] = *(const uint4*)(pB1 + k + 16);

        __builtin_prefetch(pA0 + k + 128, 0, 1);
        __builtin_prefetch(pA1 + k + 128, 0, 1);

        acc[0][0] = __builtin_amdgcn_wmma_f32_16x16x32_bf16(
            false, a0.v, false, b0.v, (short)0, acc[0][0], false, false);
        acc[0][1] = __builtin_amdgcn_wmma_f32_16x16x32_bf16(
            false, a0.v, false, b1.v, (short)0, acc[0][1], false, false);
        acc[1][0] = __builtin_amdgcn_wmma_f32_16x16x32_bf16(
            false, a1.v, false, b0.v, (short)0, acc[1][0], false, false);
        acc[1][1] = __builtin_amdgcn_wmma_f32_16x16x32_bf16(
            false, a1.v, false, b1.v, (short)0, acc[1][1], false, false);
    }

    // Epilogue. C/D layout: VGPR r holds row r (lanes 0-15) / row r+8 (lanes 16-31).
    int rb = (lane >> 4) << 3;
    for (int ni = 0; ni < 2; ni++) {
        int col = nbase + ni * 16 + (lane & 15);
        float s = scale[col], bb = bias[col];
        for (int mi = 0; mi < 2; mi++) {
            int rowb = mbase + mi * 16 + rb;
            v8f a = acc[mi][ni];
            for (int r = 0; r < 8; r++) {
                int row = rowb + r;
                if (row < M) {
                    float v = a[r] * s + bb;
                    if (act == 1)      v = fmaxf(v, 0.0f);
                    else if (act == 2) v = (v > 0.0f) ? v : 0.2f * v;
                    if (Yf) Yf[(size_t)row * ldy + col] = v;
                    if (Yb) Yb[(size_t)row * ldy + col] = f2bf(v);
                }
            }
        }
    }
}

// ---------------------------------------------------------------------------
// Max-pool over the sample dimension: out[bs, c] = max_k Y[(bs*K+k), c]
// ---------------------------------------------------------------------------
__global__ void maxpool_kernel(const float* __restrict__ Y, int Stotal, int K, int C,
                               float* __restrict__ out) {
    int t = blockIdx.x * blockDim.x + threadIdx.x;
    if (t >= Stotal * C) return;
    int c = t % C, s = t / C;
    const float* y = Y + ((size_t)s * K) * C + c;
    float m = y[0];
    for (int k = 1; k < K; k++) m = fmaxf(m, y[(size_t)k * C]);
    out[t] = m;
}

__global__ void f32_to_bf16_kernel(const float* __restrict__ in, int n,
                                   unsigned short* __restrict__ out) {
    int i = blockIdx.x * blockDim.x + threadIdx.x;
    if (i < n) out[i] = f2bf(in[i]);
}

__global__ void copy_out_kernel(const float* __restrict__ Y, int M, int Cout, int CoutPad,
                                float* __restrict__ out) {
    int i = blockIdx.x * blockDim.x + threadIdx.x;
    if (i >= M * Cout) return;
    int r = i / Cout, c = i % Cout;
    out[i] = Y[(size_t)r * CoutPad + c];
}

// ---------------------------------------------------------------------------
// Host orchestration
// ---------------------------------------------------------------------------
extern "C" void kernel_launch(void* const* d_in, const int* in_sizes, int n_in,
                              void* d_out, int out_size, void* d_ws, size_t ws_size,
                              hipStream_t stream) {
    (void)in_sizes; (void)n_in; (void)out_size; (void)ws_size;

    const int B = 16, N = 4096;
    const int S1 = 512, K1 = 32;
    const int S2 = 128, K2 = 64;
    const int K3 = 128;                 // group_all over l2's 128 points
    const int M1 = B * S1 * K1;         // 262144
    const int M2 = B * S2 * K2;         // 131072
    const int M3 = B * K3;              // 2048

    // Layer table (d_in index of 'w'; params in dict order w,b,gamma,beta,mean,var)
    static const int  Lwi[12]   = {1, 7, 13, 19, 25, 31, 37, 43, 49, 55, 61, 67};
    static const int  LCout[12] = {64, 64, 128, 128, 128, 256, 256, 512, 1024, 512, 256, 40};
    static const int  LCin[12]  = {6, 64, 64, 134, 128, 128, 259, 256, 512, 1024, 512, 256};
    static const int  LCoP[12]  = {64, 64, 128, 128, 128, 256, 256, 512, 1024, 512, 256, 64};
    static const int  LKP[12]   = {32, 64, 64, 160, 128, 128, 288, 256, 512, 1024, 512, 256};
    static const int  Lbn[12]   = {1, 1, 1, 1, 1, 1, 1, 1, 1, 1, 1, 0};

    auto fin = [&](int i) { return (const float*)d_in[i]; };

    // ---- workspace arena -------------------------------------------------
    char* ws = (char*)d_ws;
    size_t off = 0;
    auto bump = [&](size_t bytes) {
        size_t o = off;
        off += (bytes + 255) & ~(size_t)255;
        return o;
    };
    float* xyz_bn3 = (float*)(ws + bump((size_t)B * N * 3 * 4));
    int*   fps1    = (int*)(ws + bump((size_t)B * S1 * 4));
    float* nxyz1   = (float*)(ws + bump((size_t)B * S1 * 3 * 4));
    int*   bq1     = (int*)(ws + bump((size_t)B * S1 * K1 * 4));
    int*   fps2    = (int*)(ws + bump((size_t)B * S2 * 4));
    float* nxyz2   = (float*)(ws + bump((size_t)B * S2 * 3 * 4));
    int*   bq2     = (int*)(ws + bump((size_t)B * S2 * K2 * 4));
    float* pooled1 = (float*)(ws + bump((size_t)B * S1 * 128 * 4));
    float* pooled2 = (float*)(ws + bump((size_t)B * S2 * 256 * 4));
    float* pooled3 = (float*)(ws + bump((size_t)B * 1024 * 4));
    float* ybig    = (float*)(ws + bump((size_t)M1 * 128 * 4));      // 128 MB, reused
    unsigned short* sA = (unsigned short*)(ws + bump((size_t)64 << 20));
    unsigned short* sB = (unsigned short*)(ws + bump((size_t)64 << 20));
    float* yfc3    = (float*)(ws + bump((size_t)16 * 64 * 4));

    unsigned short* Wb[12];
    float *Sc[12], *Bi[12];
    for (int l = 0; l < 12; l++) {
        Wb[l] = (unsigned short*)(ws + bump((size_t)LCoP[l] * LKP[l] * 2));
        Sc[l] = (float*)(ws + bump((size_t)LCoP[l] * 4));
        Bi[l] = (float*)(ws + bump((size_t)LCoP[l] * 4));
    }

    // ---- weight / BN prep ------------------------------------------------
    for (int l = 0; l < 12; l++) {
        int tot = LCoP[l] * LKP[l];
        prep_w_kernel<<<CDIV(tot, 256), 256, 0, stream>>>(
            fin(Lwi[l]), LCout[l], LCin[l], LCoP[l], LKP[l], Wb[l]);
        const float* b = fin(Lwi[l] + 1);
        const float *g = b, *be = b, *mn = b, *vr = b;
        if (Lbn[l]) { g = fin(Lwi[l] + 2); be = fin(Lwi[l] + 3);
                      mn = fin(Lwi[l] + 4); vr = fin(Lwi[l] + 5); }
        prep_sb_kernel<<<CDIV(LCoP[l], 64), 64, 0, stream>>>(
            b, g, be, mn, vr, LCout[l], LCoP[l], Lbn[l], Sc[l], Bi[l]);
    }

    auto gemm = [&](const unsigned short* Xp, int M, int li,
                    float* Yf, unsigned short* Yb, int act) {
        dim3 grid(CDIV(M, 128), LCoP[li] / 32);
        gemm_bn_act_kernel<<<grid, 128, 0, stream>>>(
            Xp, M, LKP[li], Wb[li], Sc[li], Bi[li], Yf, Yb, LCoP[li], act);
    };

    // ---- stage 0: transpose ---------------------------------------------
    transpose_xyz_kernel<<<CDIV(B * N, 256), 256, 0, stream>>>(fin(0), B, N, xyz_bn3);

    // ---- SA1 -------------------------------------------------------------
    fps_kernel<<<B, 256, 0, stream>>>(xyz_bn3, N, S1, fps1);
    gather_xyz_kernel<<<CDIV(B * S1, 256), 256, 0, stream>>>(xyz_bn3, fps1, B, N, S1, nxyz1);
    ball_query_kernel<<<CDIV(B * S1, 256), 256, 0, stream>>>(
        xyz_bn3, nxyz1, B, N, S1, 0.2f * 0.2f, K1, bq1);
    group_kernel<<<CDIV(M1, 256), 256, 0, stream>>>(
        xyz_bn3, nxyz1, bq1, (const float*)nullptr, B, N, S1, K1, 0, LKP[0], sA);
    gemm(sA, M1, 0, nullptr, sB, 1);
    gemm(sB, M1, 1, nullptr, sA, 1);
    gemm(sA, M1, 2, ybig, nullptr, 1);
    maxpool_kernel<<<CDIV(B * S1 * 128, 256), 256, 0, stream>>>(ybig, B * S1, K1, 128, pooled1);

    // ---- SA2 -------------------------------------------------------------
    fps_kernel<<<B, 256, 0, stream>>>(nxyz1, S1, S2, fps2);
    gather_xyz_kernel<<<CDIV(B * S2, 256), 256, 0, stream>>>(nxyz1, fps2, B, S1, S2, nxyz2);
    ball_query_kernel<<<CDIV(B * S2, 256), 256, 0, stream>>>(
        nxyz1, nxyz2, B, S1, S2, 0.4f * 0.4f, K2, bq2);
    group_kernel<<<CDIV(M2, 256), 256, 0, stream>>>(
        nxyz1, nxyz2, bq2, pooled1, B, S1, S2, K2, 128, LKP[3], sB);
    gemm(sB, M2, 3, nullptr, sA, 1);
    gemm(sA, M2, 4, nullptr, sB, 1);
    gemm(sB, M2, 5, ybig, nullptr, 1);
    maxpool_kernel<<<CDIV(B * S2 * 256, 256), 256, 0, stream>>>(ybig, B * S2, K2, 256, pooled2);

    // ---- SA3 (group_all) -------------------------------------------------
    group_all_kernel<<<CDIV(M3, 256), 256, 0, stream>>>(
        nxyz2, pooled2, B, K3, 256, LKP[6], sA);
    gemm(sA, M3, 6, nullptr, sB, 1);
    gemm(sB, M3, 7, nullptr, sA, 1);
    gemm(sA, M3, 8, ybig, nullptr, 1);
    maxpool_kernel<<<CDIV(B * 1024, 256), 256, 0, stream>>>(ybig, B, K3, 1024, pooled3);

    // ---- FC head ----------------------------------------------------------
    f32_to_bf16_kernel<<<CDIV(B * 1024, 256), 256, 0, stream>>>(pooled3, B * 1024, sB);
    gemm(sB, B, 9,  nullptr, sA, 2);   // fc1: leaky 0.2
    gemm(sA, B, 10, nullptr, sB, 2);   // fc2: leaky 0.2
    gemm(sB, B, 11, yfc3, nullptr, 0); // fc3: linear
    copy_out_kernel<<<CDIV(B * 40, 64), 64, 0, stream>>>(yfc3, B, 40, 64, (float*)d_out);
}
